// PanoSelfAttention_39719857554061
// MI455X (gfx1250) — matrix operand order repaired and checked by reading, not compiled, not run on hardware
//
#include <hip/hip_runtime.h>
#include <hip/hip_bf16.h>
#include <math.h>

// Problem constants (match reference)
#define B_   4
#define H_   64
#define W_   128
#define DM_  256
#define NH_  8
#define K_   4
#define SC_  2
#define DK_  32
#define M_   (B_ * H_ * W_)          // 32768 rows
#define NOFF (2 * NH_ * K_ * SC_)    // 128
#define NATT (NH_ * K_ * SC_)        // 64
#define KT_  (DM_ / 32)              // 8 K-chunks of 32

typedef __attribute__((ext_vector_type(16))) _Float16 v16h;
typedef __attribute__((ext_vector_type(8)))  float    v8f;

union V16HU { v16h v; _Float16 h[16]; };

// ---------------------------------------------------------------------------
// Repack W (256 x N, f32 row-major) into fragment-major f16 so each lane's
// B fragment is 32 contiguous bytes:
//   Wp[ ((ntile*8 + ktile)*32 + lane)*16 + i ]
//     = (f16) W[ (ktile*32 + (lane>>4)*16 + i) * N + ntile*16 + (lane&15) ]
// in contiguous k order, matching the GEMM's A fragment (lane n = lane&15; hi lanes K+16).
// ---------------------------------------------------------------------------
__global__ void repack_w_f16(const float* __restrict__ W,
                             _Float16* __restrict__ Wp, int Ncols) {
  const long idx = (long)blockIdx.x * blockDim.x + threadIdx.x;
  const long total = (long)DM_ * Ncols;
  if (idx >= total) return;
  const int  i    = (int)(idx & 15);
  const int  L    = (int)((idx >> 4) & 31);
  const long rest = idx >> 9;            // ntile*8 + ktile
  const int  t    = (int)(rest & 7);
  const int  j    = (int)(rest >> 3);
  const int  krow = t * 32 + ((L >> 4) << 4) + i;
  const int  ncol = j * 16 + (L & 15);
  Wp[idx] = (_Float16)W[(long)krow * Ncols + ncol];
}

// ---------------------------------------------------------------------------
// C[M x N] = A[M x 256] * W[256 x N] + bias[N]   (A fp32, W pre-packed f16)
// One wave per 16x64 output strip: A fragment loaded once per K-chunk and
// reused across 4 WMMAs (4 N-tiles) -> 32 v_wmma per wave, coalesced b128
// loads on both operands.
// ---------------------------------------------------------------------------
__global__ void wmma_gemm_bias(const float* __restrict__ A,
                               const _Float16* __restrict__ Wp,
                               const float* __restrict__ bias,
                               float* __restrict__ C,
                               int Mrows, int Ncols) {
  const int lane = threadIdx.x & 31;
  const int wave = threadIdx.x >> 5;
  const int groupsN = Ncols >> 6;                    // 4 N-tiles per group
  const long numGroups = (long)(Mrows >> 4) * groupsN;
  const long g = (long)blockIdx.x * (blockDim.x >> 5) + wave;
  if (g >= numGroups) return;

  const int gn = (int)(g % groupsN);
  const int tm = (int)(g / groupsN);
  const int row0   = tm << 4;
  const int ntile0 = gn << 2;                        // first of 4 N-tiles

  const int mA    = row0 + (lane & 15);
  const int koffA = (lane >> 4) << 4;                // 0 or 16

  v8f acc[4] = {{}, {}, {}, {}};

#pragma unroll
  for (int t = 0; t < KT_; ++t) {
    // A fragment: one 64B contiguous run per lane (compiler -> b128 loads)
    V16HU af;
    const float* arow = A + (long)mA * DM_ + t * 32 + koffA;
#pragma unroll
    for (int v = 0; v < 8; ++v) {
      const int kk = v << 1;
      af.h[2 * v]     = (_Float16)arow[kk];
      af.h[2 * v + 1] = (_Float16)arow[kk + 1];
    }
    // 4 B fragments: each lane reads 32 contiguous bytes from packed weights
#pragma unroll
    for (int j = 0; j < 4; ++j) {
      const v16h bf = *(const v16h*)(Wp +
          ((((long)(ntile0 + j) * KT_ + t) * 32 + lane) << 4));
      acc[j] = __builtin_amdgcn_wmma_f32_16x16x32_f16(
          false, af.v, false, bf, (short)0, acc[j], false, false);
    }
  }

  const int mout = row0 + ((lane >> 4) << 3);
#pragma unroll
  for (int j = 0; j < 4; ++j) {
    const int nout = ((ntile0 + j) << 4) + (lane & 15);
    const float bb = bias[nout];
#pragma unroll
    for (int r = 0; r < 8; ++r)
      C[(long)(mout + r) * Ncols + nout] = acc[j][r] + bb;
  }
}

// ---------------------------------------------------------------------------
// In-place softmax over contiguous groups of 8 (one thread per (row, head)).
// ---------------------------------------------------------------------------
__global__ void softmax_groups8(float* __restrict__ p, long ngroups) {
  const long i = (long)blockIdx.x * blockDim.x + threadIdx.x;
  if (i >= ngroups) return;
  float* g = p + i * 8;
  float v[8];
  float mx = -INFINITY;
#pragma unroll
  for (int j = 0; j < 8; ++j) { v[j] = g[j]; mx = fmaxf(mx, v[j]); }
  float s = 0.f;
#pragma unroll
  for (int j = 0; j < 8; ++j) { v[j] = __expf(v[j] - mx); s += v[j]; }
  const float inv = 1.f / s;
#pragma unroll
  for (int j = 0; j < 8; ++j) g[j] = v[j] * inv;
}

// ---------------------------------------------------------------------------
// Deformable bilinear sampling + attention-weighted accumulation.
// One thread per output element (b,y,x,h,c). sf layout: (B,H,W,NH,DK).
// align_corners=False: g = p * D/(D-1) - 0.5, zero padding.
// Ref-point batch index is (b*NH+h)%B — faithful to the reference's tile().
// ---------------------------------------------------------------------------
__global__ void deform_sample(const float* __restrict__ sf0,
                              const float* __restrict__ sf1,
                              const float* __restrict__ off,
                              const float* __restrict__ Aw,
                              const float* __restrict__ ref,
                              float* __restrict__ feat) {
  const long idx = (long)blockIdx.x * blockDim.x + threadIdx.x;
  const long total = (long)M_ * DM_;
  if (idx >= total) return;

  const int  h = (int)((idx >> 5) & (NH_ - 1));
  const long m = idx >> 8;                       // row in [0, B*H*W)
  const int  x = (int)(m % W_);
  const int  y = (int)((m / W_) % H_);
  const int  b = (int)(m / ((long)W_ * H_));

  const int refb = (b * NH_ + h) % B_;           // head-major tile quirk
  const float* rp = ref + (((long)refb * H_ + y) * W_ + x) * 2;
  const float ry = rp[0], rx = rp[1];            // (y, x) pixel coords

  const float* orow = off + m * NOFF + (long)h * (SC_ * K_ * 2);
  const float* arow = Aw  + m * NATT + (long)h * (SC_ * K_);
  const long chan = idx & (DM_ - 1);             // h*DK + c

  float accv = 0.f;
#pragma unroll
  for (int l = 0; l < SC_; ++l) {
    const float* sf = l ? sf1 : sf0;
#pragma unroll
    for (int k = 0; k < K_; ++k) {
      const float oy = orow[(l * K_ + k) * 2 + 0];
      const float ox = orow[(l * K_ + k) * 2 + 1];
      const float gy = (ry + oy) * ((float)H_ / (float)(H_ - 1)) - 0.5f;
      const float gx = (rx + ox) * ((float)W_ / (float)(W_ - 1)) - 0.5f;
      const float fy0 = floorf(gy), fx0 = floorf(gx);
      const int   y0 = (int)fy0,    x0 = (int)fx0;
      const float wy1 = gy - fy0, wy0 = 1.f - wy1;
      const float wx1 = gx - fx0, wx0 = 1.f - wx1;
      float s = 0.f;
#pragma unroll
      for (int dy = 0; dy < 2; ++dy) {
#pragma unroll
        for (int dx = 0; dx < 2; ++dx) {
          const int yy = y0 + dy, xx = x0 + dx;
          if (yy >= 0 && yy < H_ && xx >= 0 && xx < W_) {
            const float wgt = (dy ? wy1 : wy0) * (dx ? wx1 : wx0);
            s += wgt * sf[(((long)b * H_ + yy) * W_ + xx) * DM_ + chan];
          }
        }
      }
      accv += arow[l * K_ + k] * s;
    }
  }
  feat[m * DM_ + chan] = accv;
}

// ---------------------------------------------------------------------------
extern "C" void kernel_launch(void* const* d_in, const int* in_sizes, int n_in,
                              void* d_out, int out_size, void* d_ws, size_t ws_size,
                              hipStream_t stream) {
  (void)in_sizes; (void)n_in; (void)out_size; (void)ws_size;

  const float* query = (const float*)d_in[0];
  const float* keys0 = (const float*)d_in[1];
  const float* keys1 = (const float*)d_in[2];
  const float* ref   = (const float*)d_in[3];
  const float* Wq = (const float*)d_in[4];   const float* bq = (const float*)d_in[5];
  const float* Wk = (const float*)d_in[6];   const float* bk = (const float*)d_in[7];
  const float* Wo = (const float*)d_in[8];   const float* bo = (const float*)d_in[9];
  const float* Wa = (const float*)d_in[10];  const float* ba = (const float*)d_in[11];
  const float* Wm = (const float*)d_in[12];  const float* bm = (const float*)d_in[13];
  float* out = (float*)d_out;

  // Workspace layout (feat aliases q: q is dead after the two q-side GEMMs)
  char* ws = (char*)d_ws;
  const size_t szQ   = (size_t)M_ * DM_  * sizeof(float);   // 32 MB
  const size_t szOff = (size_t)M_ * NOFF * sizeof(float);   // 16 MB
  const size_t szAtt = (size_t)M_ * NATT * sizeof(float);   //  8 MB
  float* q    = (float*)(ws);
  float* offb = (float*)(ws + szQ);
  float* Ab   = (float*)(ws + szQ + szOff);
  float* sf0  = (float*)(ws + szQ + szOff + szAtt);
  float* sf1  = (float*)(ws + szQ + szOff + szAtt + szQ);
  float* feat = (float*)(ws);                               // alias q
  char*  pb   = ws + szQ + szOff + szAtt + 2 * szQ;         // packed weights
  _Float16* Wq_p = (_Float16*)(pb);
  _Float16* Wk_p = (_Float16*)(pb + (size_t)DM_ * DM_  * 2);
  _Float16* Wm_p = (_Float16*)(pb + (size_t)DM_ * DM_  * 4);
  _Float16* Wo_p = (_Float16*)(pb + (size_t)DM_ * DM_  * 6);
  _Float16* Wa_p = (_Float16*)(pb + (size_t)DM_ * DM_  * 6 + (size_t)DM_ * NOFF * 2);

  const int threads = 256;                                   // 8 waves / block
  auto gemmGrid = [](long groups) { return (int)((groups + 7) / 8); };

  // 0) repack all weight matrices into fragment-major f16 (tiny, one pass)
  repack_w_f16<<<(DM_ * DM_  + 255) / 256, 256, 0, stream>>>(Wq, Wq_p, DM_);
  repack_w_f16<<<(DM_ * DM_  + 255) / 256, 256, 0, stream>>>(Wk, Wk_p, DM_);
  repack_w_f16<<<(DM_ * DM_  + 255) / 256, 256, 0, stream>>>(Wm, Wm_p, DM_);
  repack_w_f16<<<(DM_ * NOFF + 255) / 256, 256, 0, stream>>>(Wo, Wo_p, NOFF);
  repack_w_f16<<<(DM_ * NATT + 255) / 256, 256, 0, stream>>>(Wa, Wa_p, NATT);

  const long grp256 = (long)(M_ / 16) * (DM_  / 64);
  const long grp128 = (long)(M_ / 16) * (NOFF / 64);
  const long grp64  = (long)(M_ / 16) * (NATT / 64);

  // 1) q = query @ Wq + bq
  wmma_gemm_bias<<<gemmGrid(grp256), threads, 0, stream>>>(query, Wq_p, bq, q, M_, DM_);
  // 2) off = q @ Wo + bo
  wmma_gemm_bias<<<gemmGrid(grp128), threads, 0, stream>>>(q, Wo_p, bo, offb, M_, NOFF);
  // 3) A = softmax(q @ Wa + ba) per (row, head)
  wmma_gemm_bias<<<gemmGrid(grp64), threads, 0, stream>>>(q, Wa_p, ba, Ab, M_, NATT);
  softmax_groups8<<<((long)M_ * NH_ + 255) / 256, 256, 0, stream>>>(Ab, (long)M_ * NH_);
  // 4) per-level key projections
  wmma_gemm_bias<<<gemmGrid(grp256), threads, 0, stream>>>(keys0, Wk_p, bk, sf0, M_, DM_);
  wmma_gemm_bias<<<gemmGrid(grp256), threads, 0, stream>>>(keys1, Wk_p, bk, sf1, M_, DM_);
  // 5) deformable sampling + attention accumulation -> feat (B,H,W,NH,DK)
  deform_sample<<<((long)M_ * DM_ + 255) / 256, 256, 0, stream>>>(sf0, sf1, offb, Ab, ref, feat);
  // 6) out = feat @ Wm + bm
  wmma_gemm_bias<<<gemmGrid(grp256), threads, 0, stream>>>(feat, Wm_p, bm, out, M_, DM_);
}
